// GrapherModule_72241349918940
// MI455X (gfx1250) — matrix operand order repaired and checked by reading, not compiled, not run on hardware
//
#include <hip/hip_runtime.h>
#include <hip/hip_bf16.h>

typedef __attribute__((ext_vector_type(16))) _Float16 v16h;
typedef __attribute__((ext_vector_type(8)))  _Float16 v8h;
typedef __attribute__((ext_vector_type(2)))  _Float16 v2h;
typedef __attribute__((ext_vector_type(8)))  float    v8f;

#define B_    16
#define C_    256
#define N_    1024
#define HID_  512
#define KNN_  9
#define EPS_  1e-5f

// sched_group_barrier masks
#define SGB_WMMA  0x008
#define SGB_VMRD  0x020

// ---------------- WMMA fragment helpers (ISA 7.12.2 layouts) ----------------

// A fragment: 16x32 f16 tile of row-major matrix A (lda = row stride in elems).
// lanes 0-15: row M=lane, K = k0+0..7 (halfs 0-7) and k0+16..23 (halfs 8-15)
// lanes 16-31: row M=lane-16, K = k0+8..15 and k0+24..31
__device__ __forceinline__ v16h load_a_frag(const _Float16* A, int lda, int m0, int k0) {
    int lane = threadIdx.x & 31;
    int row  = m0 + (lane & 15);
    int kb   = k0 + ((lane >> 4) << 3);
    const _Float16* p = A + (size_t)row * lda + kb;
    v8h lo = *(const v8h*)(p);
    v8h hi = *(const v8h*)(p + 16);
    v16h f;
#pragma unroll
    for (int i = 0; i < 8; ++i) { f[i] = lo[i]; f[i + 8] = hi[i]; }
    return f;
}

// B fragment: 32x16 f16 tile; Bt is K-major per column: Bt[col][k], ldb = K stride.
// lanes 0-15: col N=lane, K = k0+0..7 and k0+16..23 ; lanes 16-31: col N=lane-16, K = k0+8..15 and k0+24..31
__device__ __forceinline__ v16h load_b_frag(const _Float16* Bt, int ldb, int n0, int k0) {
    int lane = threadIdx.x & 31;
    int col  = n0 + (lane & 15);
    int kb   = k0 + ((lane >> 4) << 3);
    const _Float16* p = Bt + (size_t)col * ldb + kb;
    v8h lo = *(const v8h*)(p);
    v8h hi = *(const v8h*)(p + 16);
    v16h f;
#pragma unroll
    for (int i = 0; i < 8; ++i) { f[i] = lo[i]; f[i + 8] = hi[i]; }
    return f;
}

__device__ __forceinline__ v8f wmma_f16(v16h a, v16h b, v8f c) {
    return __builtin_amdgcn_wmma_f32_16x16x32_f16(false, a, false, b, (short)0, c, false, false);
}

// Pipeline spec: loads of step s+1 issue before WMMAs of step s.
// Group order: L0, L1, W0, L2, W1, ..., L(S-1), W(S-2), W(S-1)
template <int STEPS, int NLOADS, int NW>
__device__ __forceinline__ void sched_pipeline() {
    __builtin_amdgcn_sched_group_barrier(SGB_VMRD, NLOADS, 0);
#pragma unroll
    for (int s = 1; s < STEPS; ++s) {
        __builtin_amdgcn_sched_group_barrier(SGB_VMRD, NLOADS, 0);
        __builtin_amdgcn_sched_group_barrier(SGB_WMMA, NW, 0);
    }
    __builtin_amdgcn_sched_group_barrier(SGB_WMMA, NW, 0);
}

// ---------------- prep kernels ----------------

__global__ void k_f32_to_f16(const float* __restrict__ src, _Float16* __restrict__ dst, int n) {
    int i = blockIdx.x * blockDim.x + threadIdx.x;
    if (i < n) dst[i] = (_Float16)src[i];
}

__global__ void k_bnparams(
    const float* fc1_b, const float* g1, const float* b1, const float* m1, const float* v1,
    const float* gc_b, const float* gg, const float* gb, const float* gm, const float* gv,
    const float* g2, const float* b2, const float* m2, const float* v2,
    const float* fc2_b, const float* g3, const float* b3, const float* m3, const float* v3,
    float* sc1, float* sh1, float* sc2, float* sh2, float* sc3, float* sh3) {
    int t = blockIdx.x * blockDim.x + threadIdx.x;
    if (t < C_) {
        float s = g1[t] * rsqrtf(v1[t] + EPS_);
        sc1[t] = s;
        sh1[t] = b1[t] + (fc1_b[t] - m1[t]) * s;
        float s3 = g3[t] * rsqrtf(v3[t] + EPS_);
        sc3[t] = s3;
        sh3[t] = b3[t] + (fc2_b[t] - m3[t]) * s3;
    }
    if (t < HID_) {
        float a1 = gg[t] * rsqrtf(gv[t] + EPS_);
        float a2 = g2[t] * rsqrtf(v2[t] + EPS_);
        sc2[t] = a1 * a2;
        sh2[t] = ((gc_b[t] - gm[t]) * a1 + gb[t] - m2[t]) * a2 + b2[t];
    }
}

// x (B,C,N) f32 -> XT (B,N,C) f16, LDS-tiled transpose
__global__ void k_transpose_x(const float* __restrict__ x, _Float16* __restrict__ XT) {
    __shared__ float t[32][33];
    int b = blockIdx.z;
    int n0 = blockIdx.x * 32, c0 = blockIdx.y * 32;
    int tx = threadIdx.x, ty = threadIdx.y;
#pragma unroll
    for (int i = ty; i < 32; i += 8)
        t[i][tx] = x[((size_t)b * C_ + (c0 + i)) * N_ + n0 + tx];
    __syncthreads();
#pragma unroll
    for (int i = ty; i < 32; i += 8)
        XT[((size_t)b * N_ + (n0 + i)) * C_ + c0 + tx] = (_Float16)t[tx][i];
}

// ---------------- 16x64-per-wave GEMM strip ----------------
// A row-major (lda = K), Bt k-major (ldb = K). Accumulates 4 16x16 tiles.
template <int K>
__device__ __forceinline__ void gemm_strip(const _Float16* __restrict__ A, int m0,
                                           const _Float16* __restrict__ Bt, int n0,
                                           v8f acc[4]) {
#pragma unroll
    for (int k0 = 0; k0 < K; k0 += 32) {
        v16h a = load_a_frag(A, K, m0, k0);
#pragma unroll
        for (int j = 0; j < 4; ++j) {
            v16h bf = load_b_frag(Bt, K, n0 + j * 16, k0);
            acc[j] = wmma_f16(a, bf, acc[j]);
        }
    }
    sched_pipeline<K / 32, 10, 4>();   // 10 b128 loads + 4 WMMAs per K step
}

// ---------------- fc1 GEMM: FT[n][c] = BN1(W1 @ X) ----------------
// 8 waves = 4(M) x 2(N) -> 64 x 128 block tile. grid (N/128, C/64, B).
__global__ __launch_bounds__(256) void k_fc1(
    const _Float16* __restrict__ W1h, const _Float16* __restrict__ XT,
    const float* __restrict__ sc1, const float* __restrict__ sh1,
    _Float16* __restrict__ FT) {
    int b = blockIdx.z;
    int w = threadIdx.x >> 5, lane = threadIdx.x & 31;
    int m0 = blockIdx.y * 64 + (w & 3) * 16;
    int n0 = blockIdx.x * 128 + (w >> 2) * 64;
    const _Float16* Xb = XT + (size_t)b * N_ * C_;
    v8f acc[4];
#pragma unroll
    for (int j = 0; j < 4; ++j) acc[j] = (v8f){};
    gemm_strip<C_>(W1h, m0, Xb, n0, acc);
    int hi = lane >> 4, col = lane & 15;
    int cbase = m0 + hi * 8;
#pragma unroll
    for (int j = 0; j < 4; ++j) {
        int n = n0 + j * 16 + col;
        v8h outp;
#pragma unroll
        for (int v = 0; v < 8; ++v) {
            int c = cbase + v;
            outp[v] = (_Float16)(acc[j][v] * sc1[c] + sh1[c]);
        }
        *(v8h*)(FT + ((size_t)b * N_ + n) * C_ + cbase) = outp;
    }
}

// ---------------- node squared norms ----------------
__global__ void k_norms(const _Float16* __restrict__ FT, float* __restrict__ x2) {
    int row = blockIdx.x * 8 + (threadIdx.x >> 5);   // row = b*N + n
    int lane = threadIdx.x & 31;
    v8h v = *(const v8h*)(FT + (size_t)row * C_ + lane * 8);
    float s = 0.f;
#pragma unroll
    for (int i = 0; i < 8; ++i) { float f = (float)v[i]; s += f * f; }
#pragma unroll
    for (int off = 16; off > 0; off >>= 1) s += __shfl_xor(s, off, 32);
    if (lane == 0) x2[row] = s;
}

// ---------------- fused distance strip + top-9 ----------------
// grid (N/16, B), 256 threads. LDS holds 16x1024 f32 distances.
__global__ __launch_bounds__(256) void k_knn(
    const _Float16* __restrict__ FT, const float* __restrict__ x2,
    int* __restrict__ idxOut) {
    __shared__ float distS[16 * N_];
    int b = blockIdx.y;
    int n0 = blockIdx.x * 16;
    int w = threadIdx.x >> 5, lane = threadIdx.x & 31;
    const _Float16* Fb = FT + (size_t)b * N_ * C_;
    const float* x2b = x2 + (size_t)b * N_;

    v8f acc[8];
#pragma unroll
    for (int j = 0; j < 8; ++j) acc[j] = (v8f){};
#pragma unroll
    for (int k0 = 0; k0 < C_; k0 += 32) {
        v16h a = load_a_frag(Fb, C_, n0, k0);
#pragma unroll
        for (int j = 0; j < 8; ++j) {
            v16h bf = load_b_frag(Fb, C_, (w * 8 + j) * 16, k0);
            acc[j] = wmma_f16(a, bf, acc[j]);
        }
    }
    sched_pipeline<C_ / 32, 18, 8>();  // 18 b128 loads + 8 WMMAs per K step

    int hi = lane >> 4, col = lane & 15;
#pragma unroll
    for (int j = 0; j < 8; ++j) {
        int m = (w * 8 + j) * 16 + col;
        float xm = x2b[m];
#pragma unroll
        for (int v = 0; v < 8; ++v) {
            int r = v + hi * 8;
            distS[r * N_ + m] = xm - 2.0f * acc[j][v];  // row-constant x2[n] dropped
        }
    }
    __syncthreads();

    // top-9 smallest per row; 8 waves handle 16 rows (2 each)
    for (int r = w; r < 16; r += 8) {
        unsigned mask = 0;  // per-lane taken bits for its 32 owned elements
        for (int pass = 0; pass < KNN_; ++pass) {
            float best = 3.4e38f;
            int bidx = 0x7fffffff;
            for (int i = 0; i < 32; ++i) {
                if (mask & (1u << i)) continue;
                int m = i * 32 + lane;
                float dv = distS[r * N_ + m];
                if (dv < best || (dv == best && m < bidx)) { best = dv; bidx = m; }
            }
#pragma unroll
            for (int off = 16; off > 0; off >>= 1) {
                float ov = __shfl_xor(best, off, 32);
                int oi = __shfl_xor(bidx, off, 32);
                if (ov < best || (ov == best && oi < bidx)) { best = ov; bidx = oi; }
            }
            if (lane == 0) idxOut[((size_t)b * N_ + (n0 + r)) * KNN_ + pass] = bidx;
            if ((bidx & 31) == lane) mask |= 1u << (bidx >> 5);
        }
    }
}

// ---------------- MRConv: MT[n][2c]=feat, MT[n][2c+1]=max_k(feat[idx]-feat) ----
__global__ void k_mrconv(const _Float16* __restrict__ FT, const int* __restrict__ idx,
                         _Float16* __restrict__ MT) {
    int bn = blockIdx.x;            // b*N + n
    int b = bn >> 10;
    int c = threadIdx.x;
    const _Float16* Fb = FT + (size_t)b * N_ * C_;
    const int* ip = idx + (size_t)bn * KNN_;
    float f = (float)FT[(size_t)bn * C_ + c];
    float maxd = -3.4e38f;
#pragma unroll
    for (int k = 0; k < KNN_; ++k) {
        int j = ip[k];
        float fj = (float)Fb[(size_t)j * C_ + c];
        maxd = fmaxf(maxd, fj - f);
    }
    v2h p; p[0] = (_Float16)f; p[1] = (_Float16)maxd;
    *(v2h*)(MT + (size_t)bn * (2 * C_) + 2 * c) = p;
}

// ---------------- gc GEMM + double-BN + GELU -> GT[n][h] f16 ----------------
// grid (N/128, HID/64, B)
__global__ __launch_bounds__(256) void k_gc(
    const _Float16* __restrict__ Wh, const _Float16* __restrict__ MT,
    const float* __restrict__ sc2, const float* __restrict__ sh2,
    _Float16* __restrict__ GT) {
    int b = blockIdx.z;
    int w = threadIdx.x >> 5, lane = threadIdx.x & 31;
    int m0 = blockIdx.y * 64 + (w & 3) * 16;
    int n0 = blockIdx.x * 128 + (w >> 2) * 64;
    const _Float16* Mb = MT + (size_t)b * N_ * (2 * C_);
    v8f acc[4];
#pragma unroll
    for (int j = 0; j < 4; ++j) acc[j] = (v8f){};
    gemm_strip<2 * C_>(Wh, m0, Mb, n0, acc);
    int hi = lane >> 4, col = lane & 15;
    int cbase = m0 + hi * 8;
#pragma unroll
    for (int j = 0; j < 4; ++j) {
        int n = n0 + j * 16 + col;
        v8h outp;
#pragma unroll
        for (int v = 0; v < 8; ++v) {
            int c = cbase + v;
            float y = acc[j][v] * sc2[c] + sh2[c];
            float g = 0.5f * y * (1.0f + erff(y * 0.70710678118654752f));  // exact GELU
            outp[v] = (_Float16)g;
        }
        *(v8h*)(GT + ((size_t)b * N_ + n) * HID_ + cbase) = outp;
    }
}

// ---------------- fc2 GEMM + BN3 + residual -> out f32 ----------------
// grid (N/128, C/64, B)
__global__ __launch_bounds__(256) void k_fc2(
    const _Float16* __restrict__ Wh, const _Float16* __restrict__ GT,
    const float* __restrict__ sc3, const float* __restrict__ sh3,
    const float* __restrict__ x, float* __restrict__ out) {
    int b = blockIdx.z;
    int w = threadIdx.x >> 5, lane = threadIdx.x & 31;
    int m0 = blockIdx.y * 64 + (w & 3) * 16;
    int n0 = blockIdx.x * 128 + (w >> 2) * 64;
    const _Float16* Gb = GT + (size_t)b * N_ * HID_;
    v8f acc[4];
#pragma unroll
    for (int j = 0; j < 4; ++j) acc[j] = (v8f){};
    gemm_strip<HID_>(Wh, m0, Gb, n0, acc);
    int hi = lane >> 4, col = lane & 15;
    int cbase = m0 + hi * 8;
#pragma unroll
    for (int j = 0; j < 4; ++j) {
        int n = n0 + j * 16 + col;
#pragma unroll
        for (int v = 0; v < 8; ++v) {
            int c = cbase + v;
            size_t o = ((size_t)b * C_ + c) * N_ + n;
            out[o] = acc[j][v] * sc3[c] + sh3[c] + x[o];
        }
    }
}

// ---------------- host launch ----------------

extern "C" void kernel_launch(void* const* d_in, const int* in_sizes, int n_in,
                              void* d_out, int out_size, void* d_ws, size_t ws_size,
                              hipStream_t stream) {
    const float* x     = (const float*)d_in[0];
    const float* fc1_w = (const float*)d_in[1];
    const float* fc1_b = (const float*)d_in[2];
    const float* bn1_g = (const float*)d_in[3];
    const float* bn1_b = (const float*)d_in[4];
    const float* bn1_m = (const float*)d_in[5];
    const float* bn1_v = (const float*)d_in[6];
    const float* gc_w  = (const float*)d_in[7];
    const float* gc_b  = (const float*)d_in[8];
    const float* gbn_g = (const float*)d_in[9];
    const float* gbn_b = (const float*)d_in[10];
    const float* gbn_m = (const float*)d_in[11];
    const float* gbn_v = (const float*)d_in[12];
    const float* bn2_g = (const float*)d_in[13];
    const float* bn2_b = (const float*)d_in[14];
    const float* bn2_m = (const float*)d_in[15];
    const float* bn2_v = (const float*)d_in[16];
    const float* fc2_w = (const float*)d_in[17];
    const float* fc2_b = (const float*)d_in[18];
    const float* bn3_g = (const float*)d_in[19];
    const float* bn3_b = (const float*)d_in[20];
    const float* bn3_m = (const float*)d_in[21];
    const float* bn3_v = (const float*)d_in[22];
    float* out = (float*)d_out;

    char* ws = (char*)d_ws;
    size_t off = 0;
    auto alloc = [&](size_t bytes) { char* p = ws + off; off += (bytes + 255) & ~(size_t)255; return p; };
    _Float16* W1h   = (_Float16*)alloc((size_t)C_ * C_ * 2);
    _Float16* gcWh  = (_Float16*)alloc((size_t)HID_ * 2 * C_ * 2);
    _Float16* fc2Wh = (_Float16*)alloc((size_t)C_ * HID_ * 2);
    float* sc1 = (float*)alloc(C_ * 4);   float* sh1 = (float*)alloc(C_ * 4);
    float* sc2 = (float*)alloc(HID_ * 4); float* sh2 = (float*)alloc(HID_ * 4);
    float* sc3 = (float*)alloc(C_ * 4);   float* sh3 = (float*)alloc(C_ * 4);
    _Float16* XT = (_Float16*)alloc((size_t)B_ * N_ * C_ * 2);
    _Float16* FT = (_Float16*)alloc((size_t)B_ * N_ * C_ * 2);
    float* x2 = (float*)alloc((size_t)B_ * N_ * 4);
    int* idx  = (int*)alloc((size_t)B_ * N_ * KNN_ * 4);
    _Float16* MT = (_Float16*)alloc((size_t)B_ * N_ * 2 * C_ * 2);
    _Float16* GT = (_Float16*)alloc((size_t)B_ * N_ * HID_ * 2);

    k_f32_to_f16<<<(C_ * C_ + 255) / 256, 256, 0, stream>>>(fc1_w, W1h, C_ * C_);
    k_f32_to_f16<<<(HID_ * 2 * C_ + 255) / 256, 256, 0, stream>>>(gc_w, gcWh, HID_ * 2 * C_);
    k_f32_to_f16<<<(C_ * HID_ + 255) / 256, 256, 0, stream>>>(fc2_w, fc2Wh, C_ * HID_);
    k_bnparams<<<2, 256, 0, stream>>>(fc1_b, bn1_g, bn1_b, bn1_m, bn1_v,
                                      gc_b, gbn_g, gbn_b, gbn_m, gbn_v,
                                      bn2_g, bn2_b, bn2_m, bn2_v,
                                      fc2_b, bn3_g, bn3_b, bn3_m, bn3_v,
                                      sc1, sh1, sc2, sh2, sc3, sh3);
    k_transpose_x<<<dim3(N_ / 32, C_ / 32, B_), dim3(32, 8), 0, stream>>>(x, XT);
    k_fc1<<<dim3(N_ / 128, C_ / 64, B_), 256, 0, stream>>>(W1h, XT, sc1, sh1, FT);
    k_norms<<<(B_ * N_) / 8, 256, 0, stream>>>(FT, x2);
    k_knn<<<dim3(N_ / 16, B_), 256, 0, stream>>>(FT, x2, idx);
    k_mrconv<<<B_ * N_, C_, 0, stream>>>(FT, idx, MT);
    k_gc<<<dim3(N_ / 128, HID_ / 64, B_), 256, 0, stream>>>(gcWh, MT, sc2, sh2, GT);
    k_fc2<<<dim3(N_ / 128, C_ / 64, B_), 256, 0, stream>>>(fc2Wh, GT, sc3, sh3, x, out);
}
